// MambaBlock_76098230550853
// MI455X (gfx1250) — matrix operand order repaired and checked by reading, not compiled, not run on hardware
//
#include <hip/hip_runtime.h>
#include <hip/hip_bf16.h>

typedef __attribute__((ext_vector_type(16))) _Float16 v16h;
typedef __attribute__((ext_vector_type(8)))  _Float16 v8h;
typedef __attribute__((ext_vector_type(8)))  float    v8f;

#define CCH   64
#define LSP   16384            // 128*128
#define MROWS 32768            // B*L
#define DIN   128
#define NST   16
#define HIDC  170
#define GPAD  192              // padded GDFN activation row (K mult of 32, 16B-aligned rows)
#define XPN   64               // padded x_proj N (36 -> 64)
#define PINN  352              // padded pin N (340 -> 352)
#define CHUNK 256
#define NCH   64               // LSP / CHUNK

// ---------------------------------------------------------------- weight prep (zero-padded f16)
__global__ void prep_kernel(const float* __restrict__ inw,  const float* __restrict__ xpw,
                            const float* __restrict__ pinw, const float* __restrict__ poutw,
                            const float* __restrict__ alog, const float* __restrict__ outw,
                            const float* __restrict__ attnw,
                            _Float16* inw16, _Float16* xpw16, _Float16* pinw16,
                            _Float16* poutw16, float* Aneg, _Float16* comb16) {
  int i = blockIdx.x * blockDim.x + threadIdx.x;
  if (i < 16384) { inw16[i] = (_Float16)inw[i]; return; }                       i -= 16384;
  if (i < XPN * 128) {                               // x_proj [36,128] -> [64,128]
    int n = i >> 7, k = i & 127;
    xpw16[i] = (n < 36) ? (_Float16)xpw[n * 128 + k] : (_Float16)0.0f; return;
  }                                                                              i -= XPN * 128;
  if (i < PINN * 64) {                               // pin [340,64] -> [352,64]
    int n = i >> 6, k = i & 63;
    pinw16[i] = (n < 340) ? (_Float16)pinw[n * 64 + k] : (_Float16)0.0f; return;
  }                                                                              i -= PINN * 64;
  if (i < 64 * GPAD) {                               // pout [64,170] -> [64,192]
    int n = i / GPAD, k = i % GPAD;
    poutw16[i] = (k < HIDC) ? (_Float16)poutw[n * HIDC + k] : (_Float16)0.0f; return;
  }                                                                              i -= 64 * GPAD;
  if (i < 2048) { Aneg[i] = -__expf(alog[i]); return; }                          i -= 2048;
  if (i < 8192) {                                    // fold attn_out_w @ out_proj_w -> [64,128]
    int c = i >> 7, d = i & 127;
    float s = 0.f;
    for (int e = 0; e < 64; ++e) s += attnw[c * 64 + e] * outw[e * 128 + d];
    comb16[i] = (_Float16)s;
  }
}

// ---------------------------------------------------------------- GroupNorm(1,C): 2-stage stats
__global__ void gn_partial_kernel(const float* __restrict__ x, float2* __restrict__ part) {
  int b = blockIdx.y;
  const float4* p = (const float4*)(x + (size_t)b * CCH * LSP) + (size_t)blockIdx.x * 4096;
  float s = 0.f, ss = 0.f;
  for (int i = threadIdx.x; i < 4096; i += 256) {
    float4 v = p[i];
    s  += v.x + v.y + v.z + v.w;
    ss += v.x * v.x + v.y * v.y + v.z * v.z + v.w * v.w;
  }
  __shared__ float sh[512];
  sh[threadIdx.x] = s; sh[threadIdx.x + 256] = ss;
  __syncthreads();
  for (int off = 128; off > 0; off >>= 1) {
    if ((int)threadIdx.x < off) {
      sh[threadIdx.x]       += sh[threadIdx.x + off];
      sh[threadIdx.x + 256] += sh[threadIdx.x + 256 + off];
    }
    __syncthreads();
  }
  if (threadIdx.x == 0) part[b * 64 + blockIdx.x] = make_float2(sh[0], sh[256]);
}

__global__ void gn_final_kernel(const float2* __restrict__ part, float* __restrict__ stats,
                                int statIdx) {
  int b = blockIdx.x;
  __shared__ float sh[128];
  float2 v = part[b * 64 + threadIdx.x];
  sh[threadIdx.x] = v.x; sh[64 + threadIdx.x] = v.y;
  __syncthreads();
  for (int off = 32; off > 0; off >>= 1) {
    if ((int)threadIdx.x < off) {
      sh[threadIdx.x]      += sh[threadIdx.x + off];
      sh[64 + threadIdx.x] += sh[64 + threadIdx.x + off];
    }
    __syncthreads();
  }
  if (threadIdx.x == 0) {
    const float total = (float)(CCH * LSP);
    float mean = sh[0] / total;
    float var  = sh[64] / total - mean * mean;
    stats[statIdx * 4 + b * 2 + 0] = mean;
    stats[statIdx * 4 + b * 2 + 1] = rsqrtf(var + 1e-5f);
  }
}

// normalize + NCHW -> [M,64] f16 transpose through LDS
__global__ void gn_apply_kernel(const float* __restrict__ x, const float* __restrict__ stats,
                                int statIdx, const float* __restrict__ g,
                                const float* __restrict__ bt, _Float16* __restrict__ out) {
  __shared__ float tile[64][65];
  int b  = blockIdx.y;
  int l0 = blockIdx.x * 64;
  float mean = stats[statIdx * 4 + b * 2 + 0];
  float rstd = stats[statIdx * 4 + b * 2 + 1];
  #pragma unroll
  for (int i = 0; i < 16; ++i) {
    int lin = i * 256 + threadIdx.x;
    int c = lin >> 6, l = lin & 63;
    tile[c][l] = x[((size_t)(b * CCH + c)) * LSP + l0 + l];
  }
  __syncthreads();
  #pragma unroll
  for (int i = 0; i < 16; ++i) {
    int lin = i * 256 + threadIdx.x;
    int l = lin >> 6, c = lin & 63;
    float v = (tile[c][l] - mean) * rstd * g[c] + bt[c];
    out[((size_t)(b * LSP + l0 + l)) * 64 + c] = (_Float16)v;
  }
}

// ---------------------------------------------------------------- WMMA GEMM (no load bounds:
// M mult of 32, K mult of 32, weights padded to Npad mult of 32, rows 16B-aligned)
__device__ inline v16h load_a_frag(const _Float16* __restrict__ A, int lda, int m0, int k0) {
  int lane = threadIdx.x & 31;
  const _Float16* p = A + (size_t)(m0 + (lane & 15)) * lda + k0 + ((lane >> 4) << 3);
  v8h lo = *(const v8h*)p;          // K run k0+khi .. +7
  v8h hi = *(const v8h*)(p + 16);   // K run k0+16+khi .. +7
  v16h f;
  #pragma unroll
  for (int h = 0; h < 8; ++h) { f[h] = lo[h]; f[h + 8] = hi[h]; }
  return f;
}
__device__ inline v16h load_b_frag(const _Float16* __restrict__ W, int ldb, int n0, int k0) {
  int lane = threadIdx.x & 31;
  const _Float16* p = W + (size_t)(n0 + (lane & 15)) * ldb + k0 + ((lane >> 4) << 3);
  v8h lo = *(const v8h*)p;          // K run k0+khi .. +7
  v8h hi = *(const v8h*)(p + 16);   // K run k0+16+khi .. +7
  v16h f;
  #pragma unroll
  for (int h = 0; h < 8; ++h) { f[h] = lo[h]; f[h + 8] = hi[h]; }
  return f;
}

__device__ inline void store_tile(v8f acc, float* __restrict__ Cout,
                                  const float* __restrict__ res, int m0, int n0,
                                  int N, int mode, int Cch) {
  int lane  = threadIdx.x & 31;
  int n     = n0 + (lane & 15);
  int mbase = m0 + ((lane >> 4) << 3);
  if (n >= N) return;
  if (mode == 0) {
    #pragma unroll
    for (int r = 0; r < 8; ++r) Cout[(size_t)(mbase + r) * N + n] = acc[r];
  } else {
    int b = mbase / LSP, l = mbase % LSP;           // tile never straddles batch (32 | LSP)
    size_t idx = ((size_t)(b * Cch + n)) * LSP + l; // 8 consecutive l per lane
    #pragma unroll
    for (int r = 0; r < 8; ++r) {
      float v = acc[r];
      if (res) v += res[idx + r];
      Cout[idx + r] = v;
    }
  }
}

// each wave: 32x32 macro-tile = 2 M-tiles x 2 N-tiles -> 4 WMMA per K-step
__global__ void wmma_gemm_kernel(const _Float16* __restrict__ A, const _Float16* __restrict__ W,
                                 float* __restrict__ Cout, const float* __restrict__ res,
                                 int M, int N, int K, int lda, int ldb, int mode, int Cch) {
  int wave = threadIdx.x >> 5;
  int m0   = (blockIdx.x * 4 + wave) * 32;
  int n0   = blockIdx.y * 32;
  if (m0 >= M) return;
  v8f a00, a01, a10, a11;
  #pragma unroll
  for (int i = 0; i < 8; ++i) { a00[i] = 0.f; a01[i] = 0.f; a10[i] = 0.f; a11[i] = 0.f; }
  for (int k0 = 0; k0 < K; k0 += 32) {
    if (k0 + 32 < K) {
      __builtin_prefetch(&A[(size_t)m0 * lda + k0 + 32], 0, 1);
      __builtin_prefetch(&W[(size_t)n0 * ldb + k0 + 32], 0, 1);
    }
    v16h af0 = load_a_frag(A, lda, m0,      k0);
    v16h af1 = load_a_frag(A, lda, m0 + 16, k0);
    v16h bf0 = load_b_frag(W, ldb, n0,      k0);
    v16h bf1 = load_b_frag(W, ldb, n0 + 16, k0);
    a00 = __builtin_amdgcn_wmma_f32_16x16x32_f16(false, af0, false, bf0, (short)0, a00, false, false);
    a01 = __builtin_amdgcn_wmma_f32_16x16x32_f16(false, af0, false, bf1, (short)0, a01, false, false);
    a10 = __builtin_amdgcn_wmma_f32_16x16x32_f16(false, af1, false, bf0, (short)0, a10, false, false);
    a11 = __builtin_amdgcn_wmma_f32_16x16x32_f16(false, af1, false, bf1, (short)0, a11, false, false);
  }
  store_tile(a00, Cout, res, m0,      n0,      N, mode, Cch);
  store_tile(a01, Cout, res, m0,      n0 + 16, N, mode, Cch);
  store_tile(a10, Cout, res, m0 + 16, n0,      N, mode, Cch);
  store_tile(a11, Cout, res, m0 + 16, n0 + 16, N, mode, Cch);
}

// ---------------------------------------------------------------- causal dwconv1d + SiLU
__global__ void conv_silu_kernel(const float* __restrict__ xz, const float* __restrict__ w,
                                 const float* __restrict__ bias, float* __restrict__ uf,
                                 _Float16* __restrict__ uh) {
  size_t tid = (size_t)blockIdx.x * blockDim.x + threadIdx.x;
  if (tid >= (size_t)MROWS * DIN) return;
  int d = tid & 127;
  int l = (int)((tid >> 7) & (LSP - 1));
  int b = (int)(tid >> 21);
  const float* base = xz + ((size_t)b * LSP) * 256 + d;   // u = xz[:, :128]
  float a = bias[d];
  #pragma unroll
  for (int k = 0; k < 4; ++k) {
    int ls = l - 3 + k;
    if (ls >= 0) a += w[d * 4 + k] * base[(size_t)ls * 256];
  }
  float s = a / (1.0f + __expf(-a));
  size_t o = ((size_t)b * LSP + l) * DIN + d;
  uf[o] = s;
  uh[o] = (_Float16)s;
}

// ---------------------------------------------------------------- dt = softplus(dt_r @ Wdt^T + b)
__global__ void dt_kernel(const float* __restrict__ xdbl, const float* __restrict__ dtw,
                          const float* __restrict__ dtb, float* __restrict__ dt) {
  size_t tid = (size_t)blockIdx.x * blockDim.x + threadIdx.x;
  if (tid >= (size_t)MROWS * DIN) return;
  int d = tid & 127;
  size_t m = tid >> 7;
  float a = dtb[d];
  #pragma unroll
  for (int r = 0; r < 4; ++r) a += xdbl[m * 36 + r] * dtw[d * 4 + r];
  dt[tid] = (a > 20.f) ? a : log1pf(__expf(a));
}

// ---------------------------------------------------------------- chunked selective scan
// half-wave (16 lanes) = one (b,d,chunk); lane = state index n
__global__ void scan_p1(const float* __restrict__ dt, const float* __restrict__ u,
                        const float* __restrict__ xdbl, const float* __restrict__ Aneg,
                        float* __restrict__ Pb, float* __restrict__ Qb) {
  int n = threadIdx.x & 15;
  int g = blockIdx.x * 16 + (threadIdx.x >> 4);
  int chunk = g & (NCH - 1);
  int d = (g >> 6) & (DIN - 1);
  int b = g >> 13;
  float Av = Aneg[d * NST + n];
  float P = 1.0f, Q = 0.0f;
  size_t mb = (size_t)b * LSP + (size_t)chunk * CHUNK;
  for (int t = 0; t < CHUNK; ++t) {
    size_t m = mb + t;
    float dtv = dt[m * DIN + d];
    float uv  = u[m * DIN + d];
    float Bv  = xdbl[m * 36 + 4 + n];
    float dA  = __expf(dtv * Av);
    P = P * dA;
    Q = dA * Q + dtv * uv * Bv;
  }
  size_t o = (((size_t)(b * DIN + d)) * NCH + chunk) * NST + n;
  Pb[o] = P; Qb[o] = Q;
}

__global__ void scan_p2(const float* __restrict__ Pb, const float* __restrict__ Qb,
                        float* __restrict__ Hb) {
  int tid = blockIdx.x * blockDim.x + threadIdx.x;
  if (tid >= 2 * DIN * NST) return;
  int n = tid & 15;
  int d = (tid >> 4) & (DIN - 1);
  int b = tid >> 11;
  float h = 0.0f;
  size_t base = ((size_t)(b * DIN + d)) * NCH;
  for (int c = 0; c < NCH; ++c) {
    size_t o = (base + c) * NST + n;
    Hb[o] = h;
    h = Pb[o] * h + Qb[o];
  }
}

__global__ void scan_p3(const float* __restrict__ dt, const float* __restrict__ u,
                        const float* __restrict__ xdbl, const float* __restrict__ Aneg,
                        const float* __restrict__ Hb, const float* __restrict__ Dp,
                        const float* __restrict__ xz, _Float16* __restrict__ yh) {
  int n = threadIdx.x & 15;
  int g = blockIdx.x * 16 + (threadIdx.x >> 4);
  int chunk = g & (NCH - 1);
  int d = (g >> 6) & (DIN - 1);
  int b = g >> 13;
  float Av = Aneg[d * NST + n];
  float Dv = Dp[d];
  float h  = Hb[(((size_t)(b * DIN + d)) * NCH + chunk) * NST + n];
  size_t mb = (size_t)b * LSP + (size_t)chunk * CHUNK;
  for (int t = 0; t < CHUNK; ++t) {
    size_t m = mb + t;
    float dtv = dt[m * DIN + d];
    float uv  = u[m * DIN + d];
    float Bv  = xdbl[m * 36 + 4 + n];
    float Cv  = xdbl[m * 36 + 20 + n];
    float dA  = __expf(dtv * Av);
    h = dA * h + dtv * uv * Bv;
    float part = h * Cv;
    part += __shfl_xor(part, 8, 16);
    part += __shfl_xor(part, 4, 16);
    part += __shfl_xor(part, 2, 16);
    part += __shfl_xor(part, 1, 16);
    if (n == 0) {
      float y = part + uv * Dv;
      float z = xz[m * 256 + 128 + d];          // z = xz[:, 128:]
      float gz = z / (1.0f + __expf(-z));       // silu(z)
      yh[m * DIN + d] = (_Float16)(y * gz);
    }
  }
}

// ---------------------------------------------------------------- GDFN 3x3 dwconv + GELU gate
// writes padded rows [M, GPAD] with zeros for c >= 170
__global__ void gdfn_dw_kernel(const float* __restrict__ t, const float* __restrict__ dww,
                               _Float16* __restrict__ gact) {
  size_t tid = (size_t)blockIdx.x * blockDim.x + threadIdx.x;
  if (tid >= (size_t)2 * LSP * GPAD) return;
  int c = (int)(tid % GPAD);
  size_t rest = tid / GPAD;
  int l = (int)(rest & (LSP - 1));
  int b = (int)(rest >> 14);
  size_t oidx = ((size_t)(b * LSP + l)) * GPAD + c;
  if (c >= HIDC) { gact[oidx] = (_Float16)0.0f; return; }
  int yy = l >> 7, xx = l & 127;
  const float* t1 = t + ((size_t)(b * 2 * HIDC + c)) * LSP;
  const float* t2 = t + ((size_t)(b * 2 * HIDC + c + HIDC)) * LSP;
  float a1 = 0.f, a2 = 0.f;
  #pragma unroll
  for (int ky = -1; ky <= 1; ++ky) {
    #pragma unroll
    for (int kx = -1; kx <= 1; ++kx) {
      int y2 = yy + ky, x2 = xx + kx;
      if (y2 >= 0 && y2 < 128 && x2 >= 0 && x2 < 128) {
        int ll = y2 * 128 + x2;
        int wi = (ky + 1) * 3 + (kx + 1);
        a1 += dww[c * 9 + wi] * t1[ll];
        a2 += dww[(c + HIDC) * 9 + wi] * t2[ll];
      }
    }
  }
  const float kk = 0.7978845608028654f;           // tanh-approx GELU (jax default)
  float ge = 0.5f * a1 * (1.0f + tanhf(kk * (a1 + 0.044715f * a1 * a1 * a1)));
  gact[oidx] = (_Float16)(ge * a2);
}

// ---------------------------------------------------------------- host launcher
extern "C" void kernel_launch(void* const* d_in, const int* in_sizes, int n_in,
                              void* d_out, int out_size, void* d_ws, size_t ws_size,
                              hipStream_t stream) {
  (void)in_sizes; (void)n_in; (void)out_size; (void)ws_size;
  const float* x      = (const float*)d_in[0];
  const float* gn_a_g = (const float*)d_in[1];
  const float* gn_a_b = (const float*)d_in[2];
  const float* inw    = (const float*)d_in[3];
  const float* c1w    = (const float*)d_in[4];
  const float* c1b    = (const float*)d_in[5];
  const float* xpw    = (const float*)d_in[6];
  const float* dtw    = (const float*)d_in[7];
  const float* dtbv   = (const float*)d_in[8];
  const float* alog   = (const float*)d_in[9];
  const float* Dp     = (const float*)d_in[10];
  const float* outw   = (const float*)d_in[11];
  const float* attnw  = (const float*)d_in[12];
  const float* gn2_g  = (const float*)d_in[13];
  const float* gn2_b  = (const float*)d_in[14];
  const float* pinw   = (const float*)d_in[15];
  const float* dww    = (const float*)d_in[16];
  const float* poutw  = (const float*)d_in[17];
  float* out = (float*)d_out;

  char* base = (char*)d_ws;
  size_t o = 0;
  auto alloc = [&](size_t bytes) -> char* {
    char* p = base + o;
    o += (bytes + 255) & ~(size_t)255;
    return p;
  };
  float*     stats   = (float*)alloc(64);
  float2*    part    = (float2*)alloc(128 * sizeof(float2));
  _Float16*  inw16   = (_Float16*)alloc(16384 * 2);
  _Float16*  xpw16   = (_Float16*)alloc((size_t)XPN * 128 * 2);
  _Float16*  pinw16  = (_Float16*)alloc((size_t)PINN * 64 * 2);
  _Float16*  poutw16 = (_Float16*)alloc((size_t)64 * GPAD * 2);
  _Float16*  comb16  = (_Float16*)alloc(8192 * 2);
  float*     Aneg    = (float*)alloc(2048 * 4);
  _Float16*  xs16    = (_Float16*)alloc((size_t)MROWS * 64 * 2);   // GN1 out; reused for GN2 out
  float*     xz      = (float*)alloc((size_t)MROWS * 256 * 4);
  float*     uf      = (float*)alloc((size_t)MROWS * DIN * 4);
  _Float16*  uh      = (_Float16*)alloc((size_t)MROWS * DIN * 2);  // reused as y_h
  float*     xdbl    = (float*)alloc((size_t)MROWS * 36 * 4);
  float*     dtbuf   = (float*)alloc((size_t)MROWS * DIN * 4);
  float*     Pb      = (float*)alloc((size_t)2 * DIN * NCH * NST * 4);
  float*     Qb      = (float*)alloc((size_t)2 * DIN * NCH * NST * 4);
  float*     Hb      = (float*)alloc((size_t)2 * DIN * NCH * NST * 4);
  float*     x2      = (float*)alloc((size_t)2 * CCH * LSP * 4);
  float*     tb      = (float*)alloc((size_t)2 * 2 * HIDC * LSP * 4);
  _Float16*  gact    = (_Float16*)alloc((size_t)MROWS * GPAD * 2);

  // 1. weight prep
  prep_kernel<<<272, 256, 0, stream>>>(inw, xpw, pinw, poutw, alog, outw, attnw,
                                       inw16, xpw16, pinw16, poutw16, Aneg, comb16);
  // 2. GroupNorm 1
  gn_partial_kernel<<<dim3(64, 2), 256, 0, stream>>>(x, part);
  gn_final_kernel<<<2, 64, 0, stream>>>(part, stats, 0);
  gn_apply_kernel<<<dim3(LSP / 64, 2), 256, 0, stream>>>(x, stats, 0, gn_a_g, gn_a_b, xs16);
  // 3. in_proj GEMM [32768,64]x[64,256]
  wmma_gemm_kernel<<<dim3(256, 8), 128, 0, stream>>>(xs16, inw16, xz, nullptr,
                                                     MROWS, 256, 64, 64, 64, 0, 0);
  // 4. causal dwconv1d + SiLU
  conv_silu_kernel<<<16384, 256, 0, stream>>>(xz, c1w, c1b, uf, uh);
  // 5. x_proj GEMM [32768,128]x[128,36] (N padded 64)
  wmma_gemm_kernel<<<dim3(256, 2), 128, 0, stream>>>(uh, xpw16, xdbl, nullptr,
                                                     MROWS, 36, 128, 128, 128, 0, 0);
  // 6. dt softplus
  dt_kernel<<<16384, 256, 0, stream>>>(xdbl, dtw, dtbv, dtbuf);
  // 7-9. chunked selective scan (phase3 fuses silu(z) gating -> y_h = uh)
  scan_p1<<<1024, 256, 0, stream>>>(dtbuf, uf, xdbl, Aneg, Pb, Qb);
  scan_p2<<<16, 256, 0, stream>>>(Pb, Qb, Hb);
  scan_p3<<<1024, 256, 0, stream>>>(dtbuf, uf, xdbl, Aneg, Hb, Dp, xz, uh);
  // 10. (out_proj ∘ attn_out) GEMM + residual -> x2 NCHW
  wmma_gemm_kernel<<<dim3(256, 2), 128, 0, stream>>>(uh, comb16, x2, x,
                                                     MROWS, 64, 128, 128, 128, 1, 64);
  // 11. GroupNorm 2
  gn_partial_kernel<<<dim3(64, 2), 256, 0, stream>>>(x2, part);
  gn_final_kernel<<<2, 64, 0, stream>>>(part, stats, 1);
  gn_apply_kernel<<<dim3(LSP / 64, 2), 256, 0, stream>>>(x2, stats, 1, gn2_g, gn2_b, xs16);
  // 12. pin GEMM [32768,64]x[64,340] (N padded 352) -> t NCHW
  wmma_gemm_kernel<<<dim3(256, 11), 128, 0, stream>>>(xs16, pinw16, tb, nullptr,
                                                      MROWS, 340, 64, 64, 64, 1, 340);
  // 13. 3x3 dwconv + gelu(t1)*t2 -> padded [M,192] f16
  gdfn_dw_kernel<<<24576, 256, 0, stream>>>(tb, dww, gact);
  // 14. pout GEMM [32768,192]x[192,64] + residual -> d_out NCHW
  wmma_gemm_kernel<<<dim3(256, 2), 128, 0, stream>>>(gact, poutw16, out, x2,
                                                     MROWS, 64, 192, 192, 192, 1, 64);
}
